// GATLayer_28003186770159
// MI455X (gfx1250) — matrix operand order repaired and checked by reading, not compiled, not run on hardware
//
#include <hip/hip_runtime.h>
#include <hip/hip_bf16.h>
#include <math.h>

// Problem constants (match reference)
constexpr int BB  = 2;     // batch
constexpr int NN  = 1024;  // nodes
constexpr int FIN = 128;   // input features
constexpr int HH  = 4;     // heads
constexpr int HD  = 32;    // hidden per head
constexpr int FOUT = HH * HD;  // 128

typedef __attribute__((ext_vector_type(16))) _Float16 v16h;
typedef __attribute__((ext_vector_type(8)))  _Float16 h8;
typedef __attribute__((ext_vector_type(8)))  float    v8f;
typedef __attribute__((ext_vector_type(4)))  unsigned u32x4;
typedef __attribute__((ext_vector_type(8)))  unsigned u32x8;

// ---------------------------------------------------------------------
// gfx1250 wave32 WMMA f16 fragment layouts:
//   A (16x32): lane row = lane%16; halves 0..7  -> K = ks + hi*8  + 0..7   (contiguous)
//                                halves 8..15 -> K = ks + 16 + hi*8 + 0..7 (contiguous)
//   B (32x16): lane col = lane%16; halves 0..7 -> K = ks + hi*8 + 0..7, 8..15 -> K = ks + 16 + hi*8 + 0..7 (as A)
//   C/D f32  : col = lane%16; reg r -> row = r + hi*8
// So with row-major A and K-major ("transposed") B we load fragments as
// 16-byte b128 loads.
// ---------------------------------------------------------------------

__device__ __forceinline__ v16h frag_from_2xh8(h8 lo, h8 hi) {
    v16h r;
#pragma unroll
    for (int i = 0; i < 8; ++i) { r[i] = lo[i]; r[8 + i] = hi[i]; }
    return r;
}

// =====================================================================
// Kernel 0: repack W (FIN x FOUT, f32 row-major) -> WT f16 [FOUT][FIN]
// =====================================================================
__global__ __launch_bounds__(128)
void k_repackW(const float* __restrict__ W, _Float16* __restrict__ WT) {
    const int c = blockIdx.x;      // 0..FOUT-1
    const int k = threadIdx.x;     // 0..FIN-1
    WT[(size_t)c * FIN + k] = (_Float16)W[(size_t)k * FOUT + c];
}

// =====================================================================
// Kernel 1: projection  x = node_feat @ W.
// One wave -> 16x32 output tile (2 accumulators share one A fragment).
// A from nf (f32 row-major, cvt to f16), B from WT (f16 K-contiguous).
// Stores xf f32 [bh][n][d] (for scores) and xT f16 [bh][d][n] (for aggregation).
// =====================================================================
__global__ __launch_bounds__(32)
void k_project(const float* __restrict__ nf, const _Float16* __restrict__ WT,
               float* __restrict__ xf, _Float16* __restrict__ xT) {
    const int lane   = threadIdx.x & 31;
    const int lane16 = lane & 15;
    const int hi     = lane >> 4;
    const int mt     = blockIdx.x;              // 0..127 row tile over B*N
    const int ct0    = blockIdx.y * 2;          // col tiles {ct0, ct0+1}
    const int arow   = mt * 16 + lane16;

    const float* arp = nf + (size_t)arow * FIN;
    const _Float16* b0p = WT + (size_t)(ct0 * 16 + lane16) * FIN;
    const _Float16* b1p = WT + (size_t)((ct0 + 1) * 16 + lane16) * FIN;

    v8f acc0 = {}, acc1 = {};
    for (int ks = 0; ks < FIN; ks += 32) {
        // A fragment: convert 16 f32 -> f16 (two contiguous float4-pairs)
        const float4* pa0 = (const float4*)(arp + ks + hi * 8);
        const float4* pa1 = (const float4*)(arp + ks + 16 + hi * 8);
        float4 f0 = pa0[0], f1 = pa0[1], f2 = pa1[0], f3 = pa1[1];
        v16h a;
        a[0]=(_Float16)f0.x; a[1]=(_Float16)f0.y; a[2]=(_Float16)f0.z; a[3]=(_Float16)f0.w;
        a[4]=(_Float16)f1.x; a[5]=(_Float16)f1.y; a[6]=(_Float16)f1.z; a[7]=(_Float16)f1.w;
        a[8]=(_Float16)f2.x; a[9]=(_Float16)f2.y; a[10]=(_Float16)f2.z; a[11]=(_Float16)f2.w;
        a[12]=(_Float16)f3.x; a[13]=(_Float16)f3.y; a[14]=(_Float16)f3.z; a[15]=(_Float16)f3.w;

        const h8* pb0 = (const h8*)(b0p + ks + hi * 8);
        const h8* pb1 = (const h8*)(b1p + ks + hi * 8);
        v16h b0 = frag_from_2xh8(pb0[0], pb0[2]);
        v16h b1 = frag_from_2xh8(pb1[0], pb1[2]);

        acc0 = __builtin_amdgcn_wmma_f32_16x16x32_f16(false, a, false, b0, (short)0, acc0, false, false);
        acc1 = __builtin_amdgcn_wmma_f32_16x16x32_f16(false, a, false, b1, (short)0, acc1, false, false);
    }

#pragma unroll
    for (int r = 0; r < 8; ++r) {
        const int m  = mt * 16 + r + (hi ? 8 : 0);  // global row in B*N
        const int b_ = m >> 10;                     // /NN
        const int n  = m & (NN - 1);
#pragma unroll
        for (int t = 0; t < 2; ++t) {
            const int c = (ct0 + t) * 16 + lane16;
            const int h = c >> 5;                   // /HD
            const int d = c & (HD - 1);
            const int bh = b_ * HH + h;
            const float v = (t == 0) ? acc0[r] : acc1[r];
            xf[((size_t)bh * NN + n) * HD + d] = v;
            xT[((size_t)bh * HD + d) * NN + n] = (_Float16)v;
        }
    }
}

// =====================================================================
// Kernel 2: pairwise scores with mask.
// e[b,h,i,j] = adj ? sum_k Wa[h,k]*leaky(x_i[k]+x_j[k]) : -1e30
// leaky(v,0.2) == 0.6*v + 0.4*|v|
// x_i tile (16x32 f32) staged into LDS by the Tensor Data Mover.
// =====================================================================
__global__ __launch_bounds__(256)
void k_scores(const float* __restrict__ xf, const float* __restrict__ Wa,
              const int* __restrict__ adj, float* __restrict__ e) {
    const int blk = blockIdx.x;
    const int i0  = (blk % (NN / 16)) * 16;
    const int bh  = blk / (NN / 16);           // 0..B*H-1
    const int b_  = bh / HH;
    const int h   = bh % HH;

    extern __shared__ float smem[];            // dynamic-only LDS -> base offset 0
    float (*sxi)[HD] = (float (*)[HD])smem;    // 16 x 32 f32 = 2048 B at offset 0
    float* swa = smem + 16 * HD;               // 32 f32

    // --- TDM: async-copy the 16x32 f32 x_i tile into LDS (wave 0 issues) ---
    if (threadIdx.x < 32) {
        const unsigned long long ga =
            (unsigned long long)(xf + ((size_t)bh * NN + i0) * HD);  // tile start, bytes
        // D# group 0: count=1 | lds_addr=0 | global_addr | type=2 (bits 127:126)
        u32x4 g0 = { 1u, 0u, (unsigned)ga,
                     (unsigned)((ga >> 32) & 0x01FFFFFFull) | 0x80000000u };
        // D# group 1: data_size=4B(code 2); tensor_dim0=32; tensor_dim1=1024;
        //             tile_dim0=32; tile_dim1=16; tensor_dim0_stride=32
        u32x8 g1 = { 2u << 16,      // [17:16] data_size
                     32u << 16,     // [31:16] tensor_dim0 lo16
                     1024u << 16,   // [15:0] dim0 hi16=0 | [31:16] tensor_dim1 lo16
                     32u << 16,     // [15:0] dim1 hi16=0 | [31:16] tile_dim0
                     16u,           // [15:0] tile_dim1 | [31:16] tile_dim2=0
                     32u,           // tensor_dim0_stride lo32
                     0u, 0u };
        asm volatile("tensor_load_to_lds %0, %1" :: "s"(g0), "s"(g1) : "memory");
        __builtin_amdgcn_s_wait_tensorcnt(0);
    }
    if (threadIdx.x < HD) swa[threadIdx.x] = Wa[h * HD + threadIdx.x];
    __syncthreads();

    const int wave = threadIdx.x >> 5;
    const int lane = threadIdx.x & 31;
    const float* xjbase = xf + (size_t)bh * NN * HD;

    for (int rr = 0; rr < 2; ++rr) {
        const int il = wave * 2 + rr;          // 0..15
        const int ig = i0 + il;
        const int* adjrow = adj + ((size_t)b_ * NN + ig) * NN;
        float xi[HD];
#pragma unroll
        for (int k = 0; k < HD; ++k) xi[k] = sxi[il][k];

        float* erow = e + ((size_t)bh * NN + ig) * NN;
        for (int j = lane; j < NN; j += 32) {
            if (j + 32 < NN)
                __builtin_prefetch(xjbase + (size_t)(j + 32) * HD, 0, 0);
            const float4* xj4 = (const float4*)(xjbase + (size_t)j * HD);
            float s = 0.0f;
#pragma unroll
            for (int q = 0; q < HD / 4; ++q) {
                const float4 x4 = xj4[q];
                float v;
                v = xi[4*q+0] + x4.x; s = fmaf(swa[4*q+0], fmaf(0.4f, fabsf(v), 0.6f*v), s);
                v = xi[4*q+1] + x4.y; s = fmaf(swa[4*q+1], fmaf(0.4f, fabsf(v), 0.6f*v), s);
                v = xi[4*q+2] + x4.z; s = fmaf(swa[4*q+2], fmaf(0.4f, fabsf(v), 0.6f*v), s);
                v = xi[4*q+3] + x4.w; s = fmaf(swa[4*q+3], fmaf(0.4f, fabsf(v), 0.6f*v), s);
            }
            erow[j] = (adjrow[j] > 0) ? s : -1e30f;
        }
    }
}

// =====================================================================
// Kernel 3: row softmax over j (one block per (b,h,i) row), output f16 att.
// =====================================================================
__global__ __launch_bounds__(256)
void k_softmax(const float* __restrict__ e, _Float16* __restrict__ att) {
    const int row = blockIdx.x;                // 0..B*H*N-1
    const float* er = e + (size_t)row * NN;
    __shared__ float red[256];

    float m = -1e30f;
    for (int j = threadIdx.x; j < NN; j += blockDim.x) m = fmaxf(m, er[j]);
    red[threadIdx.x] = m;
    __syncthreads();
    for (int s = 128; s > 0; s >>= 1) {
        if (threadIdx.x < s)
            red[threadIdx.x] = fmaxf(red[threadIdx.x], red[threadIdx.x + s]);
        __syncthreads();
    }
    m = red[0];
    __syncthreads();

    float sum = 0.0f;
    for (int j = threadIdx.x; j < NN; j += blockDim.x) sum += __expf(er[j] - m);
    red[threadIdx.x] = sum;
    __syncthreads();
    for (int s = 128; s > 0; s >>= 1) {
        if (threadIdx.x < s) red[threadIdx.x] += red[threadIdx.x + s];
        __syncthreads();
    }
    const float inv = 1.0f / red[0];

    _Float16* ar = att + (size_t)row * NN;
    for (int j = threadIdx.x; j < NN; j += blockDim.x)
        ar[j] = (_Float16)(__expf(er[j] - m) * inv);
}

// =====================================================================
// Kernel 4: aggregation  xp = att @ x  per (b,h).
// One wave -> 16x32 output tile (full HD), K=1024 in 32 WMMA steps.
// All fragments are contiguous b128 loads (att row-major, xT d-major).
// Writes directly into concat-head output layout (B, N, H*HD).
// =====================================================================
__global__ __launch_bounds__(32)
void k_aggregate(const _Float16* __restrict__ att, const _Float16* __restrict__ xT,
                 float* __restrict__ out) {
    int blk = blockIdx.x;
    const int mt = blk % (NN / 16);    blk /= (NN / 16); // i-tile: 0..63
    const int bh = blk;                                  // 0..B*H-1
    const int b_ = bh / HH;
    const int h  = bh % HH;

    const int lane   = threadIdx.x & 31;
    const int lane16 = lane & 15;
    const int hi     = lane >> 4;

    const int irow = mt * 16 + lane16;
    const _Float16* arp = att + ((size_t)bh * NN + irow) * NN;
    const _Float16* x0p = xT + ((size_t)bh * HD + lane16) * NN;       // d = lane16
    const _Float16* x1p = xT + ((size_t)bh * HD + 16 + lane16) * NN;  // d = 16+lane16

    v8f acc0 = {}, acc1 = {};
    for (int ks = 0; ks < NN; ks += 32) {
        const h8* pa0 = (const h8*)(arp + ks + hi * 8);
        const h8* pa1 = (const h8*)(arp + ks + 16 + hi * 8);
        v16h a = frag_from_2xh8(pa0[0], pa1[0]);

        const h8* pb0 = (const h8*)(x0p + ks + hi * 8);
        const h8* pb1 = (const h8*)(x1p + ks + hi * 8);
        v16h b0 = frag_from_2xh8(pb0[0], pb0[2]);
        v16h b1 = frag_from_2xh8(pb1[0], pb1[2]);

        acc0 = __builtin_amdgcn_wmma_f32_16x16x32_f16(false, a, false, b0, (short)0, acc0, false, false);
        acc1 = __builtin_amdgcn_wmma_f32_16x16x32_f16(false, a, false, b1, (short)0, acc1, false, false);
    }

#pragma unroll
    for (int r = 0; r < 8; ++r) {
        const int i = mt * 16 + r + (hi ? 8 : 0);
        float* orow = out + ((size_t)b_ * NN + i) * FOUT + h * HD;
        orow[lane16]      = acc0[r];
        orow[16 + lane16] = acc1[r];
    }
}

// =====================================================================
extern "C" void kernel_launch(void* const* d_in, const int* in_sizes, int n_in,
                              void* d_out, int out_size, void* d_ws, size_t ws_size,
                              hipStream_t stream) {
    const float* nf  = (const float*)d_in[0];   // (B, N, FIN)
    const int*   adj = (const int*)  d_in[1];   // (B, N, N)
    const float* W   = (const float*)d_in[2];   // (FIN, H*HD)
    const float* Wa  = (const float*)d_in[3];   // (H, HD)
    float* out = (float*)d_out;                 // (B, N, H*HD)

    char* ws = (char*)d_ws;
    // workspace layout (bytes):
    //   xf  f32 [B*H][N][HD]    : 1 MB     @ 0
    //   xT  f16 [B*H][HD][N]    : 512 KB   @ 1 MB
    //   WT  f16 [FOUT][FIN]     : 32 KB    @ 1.5 MB
    //   att f16 [B*H][N][N]     : 16 MB    @ 2 MB
    //   e   f32 [B*H][N][N]     : 32 MB    @ 18 MB
    float*    xf  = (float*)   (ws);
    _Float16* xT  = (_Float16*)(ws + ((size_t)1 << 20));
    _Float16* WT  = (_Float16*)(ws + ((size_t)3 << 19));
    _Float16* att = (_Float16*)(ws + ((size_t)2 << 20));
    float*    e   = (float*)   (ws + ((size_t)18 << 20));

    // 0) transpose + f16 weights
    k_repackW<<<FOUT, FIN, 0, stream>>>(W, WT);

    // 1) projection (WMMA, 16x32 tile per wave)
    dim3 g1((BB * NN) / 16, FOUT / 32);
    k_project<<<g1, 32, 0, stream>>>(nf, WT, xf, xT);

    // 2) pairwise GATv2 scores + mask (TDM stages x_i tile into LDS)
    const int smem = (16 * HD + HD) * sizeof(float);
    k_scores<<<BB * HH * (NN / 16), 256, smem, stream>>>(xf, Wa, adj, e);

    // 3) row softmax -> f16 attention
    k_softmax<<<BB * HH * NN, 256, 0, stream>>>(e, att);

    // 4) aggregation (WMMA, 16x32 tile per wave)
    k_aggregate<<<BB * HH * (NN / 16), 32, 0, stream>>>(att, xT, out);
}
